// MultiHeadAttention_21036749816102
// MI455X (gfx1250) — matrix operand order repaired and checked by reading, not compiled, not run on hardware
//
#include <hip/hip_runtime.h>
#include <hip/hip_bf16.h>

// ---------------------------------------------------------------------------
// MultiHeadAttention forward for gfx1250 (MI455X): bf16 WMMA GEMMs + flash attn
// Round 3: double-buffered LDS with async global->LDS DMA overlapped under the
// WMMAs (one barrier per k-step), constant-trip staging loops (no exec-mask
// guards), 64x64 wave tiles (16 wmma/k-step/wave).
// ---------------------------------------------------------------------------

#ifndef USE_ASYNC_LDS
#define USE_ASYNC_LDS 1
#endif

typedef __attribute__((ext_vector_type(16))) __bf16 v16bf;
typedef __attribute__((ext_vector_type(8)))  __bf16 v8bf;
typedef __attribute__((ext_vector_type(8)))  float  v8f;

#define MHA_B 4
#define MHA_S 2048
#define MHA_E 1024
#define MHA_H 16
#define MHA_D 64

__device__ __forceinline__ v16bf make_v16(v8bf a, v8bf b) {
  return __builtin_shufflevector(a, b, 0,1,2,3,4,5,6,7,8,9,10,11,12,13,14,15);
}

// Async copy of one 16B chunk: global -> LDS, tracked by ASYNCcnt.
__device__ __forceinline__ void async_b128(void* lds_dst, const void* gsrc) {
#if USE_ASYNC_LDS
  unsigned la = (unsigned)(size_t)lds_dst;               // low 32 bits = LDS offset
  unsigned long long ga = (unsigned long long)(size_t)gsrc;
  asm volatile("global_load_async_to_lds_b128 %0, %1, off"
               :: "v"(la), "v"(ga) : "memory");
#else
  *(uint4*)lds_dst = *(const uint4*)gsrc;
#endif
}

__device__ __forceinline__ void async_wait() {
#if USE_ASYNC_LDS
  asm volatile("s_wait_asynccnt 0x0" ::: "memory");
#endif
}

// ---------------------------------------------------------------------------
// fp32 -> bf16 conversion
// ---------------------------------------------------------------------------
__global__ void cvt_f32_bf16(const float* __restrict__ in, __bf16* __restrict__ out, int n) {
  int i = blockIdx.x * blockDim.x + threadIdx.x;
  if (i < n) out[i] = (__bf16)in[i];
}

// ---------------------------------------------------------------------------
// GEMM: out[m,n] = sum_k A[m,k] * W[n,k] + bias[n]
// A: [M,K] bf16 row-major, W: [N,K] bf16 row-major.
// mode 0: f32 row-major [M,N] -> outf ; mode 1: bf16 scatter to [B,H,S,D].
// Block: 128 threads (4 waves). Tile: BM=128, BN=128, BK=32, double-buffered.
// Waves 2(M) x 2(N); each wave owns 64x64 (4x4 WMMA accums, 16 wmma/k-step).
// Pipeline per k-step: wait(tile t) -> barrier -> async-issue(tile t+1) ->
//                      16 WMMA on tile t   (DMA overlaps compute)
// ---------------------------------------------------------------------------
#define GBM 128
#define GBN 128
#define GBK 32
#define GPITCH 40   // BK + 8 pad; row stride 80B (16B multiple)

__global__ __launch_bounds__(128)
void gemm_bf16(const __bf16* __restrict__ A, const __bf16* __restrict__ W,
               const float* __restrict__ bias, float* __restrict__ outf,
               __bf16* __restrict__ outb, int M, int N, int K, int mode)
{
  __shared__ __align__(16) __bf16 As[2][GBM][GPITCH];
  __shared__ __align__(16) __bf16 Ws[2][GBN][GPITCH];

  const int tid  = threadIdx.x;
  const int lane = tid & 31;
  const int w    = tid >> 5;
  const int lo   = (lane >> 4) & 1;   // lane half
  const int ln   = lane & 15;         // A: M row / B,C: N col
  const int wm   = w & 1;             // wave M index 0..1
  const int wn   = w >> 1;            // wave N index 0..1
  const int m0   = blockIdx.y * GBM;
  const int n0   = blockIdx.x * GBN;

  // Stage one 128x32 A tile + 128x32 W tile into buffer `buf` (async DMA).
  auto stage = [&](int kt, int buf) {
    const int k0 = kt * GBK;
#pragma unroll
    for (int t = 0; t < 4; ++t) {           // constant trip: no exec guards
      int c   = tid + t * 128;
      int row = c >> 2, cc = c & 3;
      async_b128(&As[buf][row][cc * 8], &A[(size_t)(m0 + row) * K + k0 + cc * 8]);
      async_b128(&Ws[buf][row][cc * 8], &W[(size_t)(n0 + row) * K + k0 + cc * 8]);
    }
  };

  v8f acc[4][4];
#pragma unroll
  for (int j = 0; j < 4; ++j) {
    float bv = bias[n0 + wn * 64 + j * 16 + ln];
#pragma unroll
    for (int i = 0; i < 4; ++i)
#pragma unroll
      for (int r = 0; r < 8; ++r) acc[i][j][r] = bv;
  }

  const int KT = K / GBK;
  stage(0, 0);

  for (int kt = 0; kt < KT; ++kt) {
    async_wait();          // tile kt landed in LDS
    __syncthreads();       // visible to all waves; prior buffer free for reuse
    if (kt + 1 < KT) {
      stage(kt + 1, (kt + 1) & 1);   // DMA next tile under this tile's WMMAs
      __builtin_prefetch(&A[(size_t)(m0 + tid) * K + (kt + 1) * GBK], 0, 0);
      __builtin_prefetch(&W[(size_t)(n0 + tid) * K + (kt + 1) * GBK], 0, 0);
    }
    const int buf = kt & 1;

    v16bf af[4], bfr[4];
#pragma unroll
    for (int i = 0; i < 4; ++i) {
      const __bf16* ap = &As[buf][wm * 64 + i * 16 + ln][0];
      v8bf a0 = *(const v8bf*)(ap + lo * 8);        // K = lo*8 .. +7
      v8bf a1 = *(const v8bf*)(ap + 16 + lo * 8);   // K = 16+lo*8 .. +7
      af[i] = make_v16(a0, a1);
    }
#pragma unroll
    for (int j = 0; j < 4; ++j) {
      const __bf16* bp = &Ws[buf][wn * 64 + j * 16 + ln][0];
      v8bf b0 = *(const v8bf*)(bp + lo * 8);        // K = lo*8 .. +7
      v8bf b1 = *(const v8bf*)(bp + 16 + lo * 8);   // K = 16+lo*8 .. +7
      bfr[j] = make_v16(b0, b1);
    }
#pragma unroll
    for (int i = 0; i < 4; ++i)
#pragma unroll
      for (int j = 0; j < 4; ++j)
        acc[i][j] = __builtin_amdgcn_wmma_f32_16x16x32_bf16(
            false, af[i], false, bfr[j], (short)0, acc[i][j], false, false);
  }

  // Epilogue. C/D layout: lane L -> (m = r + 8*(L/16), n = L%16)
#pragma unroll
  for (int i = 0; i < 4; ++i)
#pragma unroll
    for (int j = 0; j < 4; ++j) {
      int n = n0 + wn * 64 + j * 16 + ln;
#pragma unroll
      for (int r = 0; r < 8; ++r) {
        int m = m0 + wm * 64 + i * 16 + r + 8 * lo;
        if (mode == 0) {
          outf[(size_t)m * N + n] = acc[i][j][r];
        } else {
          int b = m >> 11, s = m & (MHA_S - 1);
          int h = n >> 6, d = n & (MHA_D - 1);
          outb[((((size_t)b * MHA_H + h) * MHA_S) + s) * MHA_D + d] =
              (__bf16)acc[i][j][r];
        }
      }
    }
}

// ---------------------------------------------------------------------------
// Flash attention. Q,K,V: [B,H,S,D] bf16. ctx out: [B,S,H,D] bf16.
// Grid: B*H*(S/64) blocks, 128 threads (4 waves), wave w owns 16 query rows.
// K/V tiles double-buffered; K via async LDS DMA, V transposed through VGPRs.
// ---------------------------------------------------------------------------
__global__ __launch_bounds__(128)
void flash_attn(const __bf16* __restrict__ Q, const __bf16* __restrict__ Kg,
                const __bf16* __restrict__ Vg, __bf16* __restrict__ ctx)
{
  __shared__ __align__(16) __bf16 Ks[2][64][72];     // [key_local][d]
  __shared__ __align__(16) __bf16 Vs[2][64][72];     // transposed: [d][key_local]
  __shared__ __align__(16) __bf16 Ps[4][16][72];     // per-wave P stage

  const int tid  = threadIdx.x;
  const int lane = tid & 31;
  const int w    = tid >> 5;
  const int lo   = (lane >> 4) & 1;
  const int ln   = lane & 15;

  const int blk = blockIdx.x;
  const int qt  = blk & 31;        // S/64 = 32 query tiles
  const int bh  = blk >> 5;        // b*H + h
  const size_t base = (size_t)bh * MHA_S * MHA_D;
  const int qRow0 = qt * 64 + w * 16;
  const int NKB = MHA_S / 64;      // 32 key blocks

  // Stage key block kb into buffer `buf`.
  auto stageKV = [&](int kb, int buf) {
    const __bf16* kp = Kg + base + (size_t)kb * 64 * MHA_D;
    const __bf16* vp = Vg + base + (size_t)kb * 64 * MHA_D;
#pragma unroll
    for (int t = 0; t < 4; ++t) {           // constant trip: no exec guards
      int c   = tid + t * 128;
      int row = c >> 3, cc = c & 7;
      async_b128(&Ks[buf][row][cc * 8], &kp[row * MHA_D + cc * 8]);
    }
#pragma unroll
    for (int t = 0; t < 4; ++t) {
      int c   = tid + t * 128;
      int row = c >> 3, cc = c & 7;
      v8bf v = *(const v8bf*)&vp[row * MHA_D + cc * 8];
#pragma unroll
      for (int e = 0; e < 8; ++e) Vs[buf][cc * 8 + e][row] = v[e];
    }
  };

  // Q A-fragments (persist across key loop): K-dim = head dim, split 0..31 / 32..63
  v16bf qf[2];
  {
    const __bf16* qp = Q + base + (size_t)(qRow0 + ln) * MHA_D;
#pragma unroll
    for (int kk = 0; kk < 2; ++kk) {
      v8bf q0 = *(const v8bf*)(qp + kk * 32 + lo * 8);
      v8bf q1 = *(const v8bf*)(qp + kk * 32 + lo * 8 + 16);
      qf[kk] = make_v16(q0, q1);
    }
  }

  v8f o[4];
#pragma unroll
  for (int j = 0; j < 4; ++j)
#pragma unroll
    for (int r = 0; r < 8; ++r) o[j][r] = 0.f;

  float mrow[8], lrow[8];
#pragma unroll
  for (int r = 0; r < 8; ++r) { mrow[r] = -1e30f; lrow[r] = 0.f; }

  const float scale = 0.125f;  // 1/sqrt(64)

  stageKV(0, 0);

  for (int kb = 0; kb < NKB; ++kb) {
    async_wait();          // K tile kb landed in LDS
    __syncthreads();       // V/K visible to all; previous buffer free
    if (kb + 1 < NKB) {
      stageKV(kb + 1, (kb + 1) & 1);   // overlap next tile DMA with compute
      __builtin_prefetch(&Kg[base + (size_t)(kb + 2) * 64 * MHA_D + tid * 32], 0, 0);
      __builtin_prefetch(&Vg[base + (size_t)(kb + 2) * 64 * MHA_D + tid * 32], 0, 0);
    }
    const int buf = kb & 1;

    // Scores: S = Q * K^T  (4 x 16-wide key tiles, K-dim 64 = 2 WMMA steps)
    v8f sc[4];
#pragma unroll
    for (int j = 0; j < 4; ++j)
#pragma unroll
      for (int r = 0; r < 8; ++r) sc[j][r] = 0.f;
#pragma unroll
    for (int j = 0; j < 4; ++j) {
      const __bf16* bp = &Ks[buf][j * 16 + ln][0];
#pragma unroll
      for (int kk = 0; kk < 2; ++kk) {
        v8bf b0 = *(const v8bf*)(bp + kk * 32 + lo * 8);
        v8bf b1 = *(const v8bf*)(bp + kk * 32 + lo * 8 + 16);
        v16bf kfr = make_v16(b0, b1);
        sc[j] = __builtin_amdgcn_wmma_f32_16x16x32_bf16(
            false, qf[kk], false, kfr, (short)0, sc[j], false, false);
      }
    }
#pragma unroll
    for (int j = 0; j < 4; ++j)
#pragma unroll
      for (int r = 0; r < 8; ++r) sc[j][r] *= scale;

    // Online softmax per row (row lives across 16 lanes of one half-wave)
    float alpha[8];
#pragma unroll
    for (int r = 0; r < 8; ++r) {
      float lm = fmaxf(fmaxf(sc[0][r], sc[1][r]), fmaxf(sc[2][r], sc[3][r]));
#pragma unroll
      for (int off = 8; off >= 1; off >>= 1)
        lm = fmaxf(lm, __shfl_xor(lm, off, 16));
      float mnew = fmaxf(mrow[r], lm);
      alpha[r] = __expf(mrow[r] - mnew);
      mrow[r] = mnew;
      float rs = 0.f;
#pragma unroll
      for (int j = 0; j < 4; ++j) {
        float p = __expf(sc[j][r] - mnew);
        sc[j][r] = p;
        rs += p;
      }
#pragma unroll
      for (int off = 8; off >= 1; off >>= 1)
        rs += __shfl_xor(rs, off, 16);
      lrow[r] = alpha[r] * lrow[r] + rs;
#pragma unroll
      for (int j = 0; j < 4; ++j) o[j][r] *= alpha[r];
    }

    // Stage P through per-wave LDS to convert C-layout -> A-layout
#pragma unroll
    for (int j = 0; j < 4; ++j)
#pragma unroll
      for (int r = 0; r < 8; ++r)
        Ps[w][r + 8 * lo][j * 16 + ln] = (__bf16)sc[j][r];

    // O += P * V  (K-dim = 64 keys = 2 WMMA steps; B from transposed Vs)
#pragma unroll
    for (int kk = 0; kk < 2; ++kk) {
      const __bf16* pp = &Ps[w][ln][0];
      v8bf p0 = *(const v8bf*)(pp + kk * 32 + lo * 8);
      v8bf p1 = *(const v8bf*)(pp + kk * 32 + lo * 8 + 16);
      v16bf pf = make_v16(p0, p1);
#pragma unroll
      for (int j = 0; j < 4; ++j) {
        const __bf16* vb = &Vs[buf][j * 16 + ln][0];
        v8bf b0 = *(const v8bf*)(vb + kk * 32 + lo * 8);
        v8bf b1 = *(const v8bf*)(vb + kk * 32 + lo * 8 + 16);
        v16bf vf = make_v16(b0, b1);
        o[j] = __builtin_amdgcn_wmma_f32_16x16x32_bf16(
            false, pf, false, vf, (short)0, o[j], false, false);
      }
    }
  }

  // Write context as [B, S, H, D] so output GEMM K-index == h*64+d
  const int b = bh >> 4, h = bh & (MHA_H - 1);
#pragma unroll
  for (int j = 0; j < 4; ++j)
#pragma unroll
    for (int r = 0; r < 8; ++r) {
      int s = qRow0 + r + 8 * lo;
      int d = j * 16 + ln;
      ctx[(((size_t)b * MHA_S + s) * MHA_H + h) * MHA_D + d] =
          (__bf16)(o[j][r] / lrow[r]);
    }
}

// ---------------------------------------------------------------------------
// Launcher
// ---------------------------------------------------------------------------
extern "C" void kernel_launch(void* const* d_in, const int* in_sizes, int n_in,
                              void* d_out, int out_size, void* d_ws, size_t ws_size,
                              hipStream_t stream) {
  const float* x  = (const float*)d_in[0];
  const float* Wq = (const float*)d_in[1];
  const float* bq = (const float*)d_in[2];
  const float* Wk = (const float*)d_in[3];
  const float* bk = (const float*)d_in[4];
  const float* Wv = (const float*)d_in[5];
  const float* bv = (const float*)d_in[6];
  const float* Wo = (const float*)d_in[7];
  const float* bo = (const float*)d_in[8];
  float* out = (float*)d_out;

  const int M = MHA_B * MHA_S;     // 8192
  const int E = MHA_E;             // 1024
  const int nX = M * E;            // 8388608
  const int nW = E * E;            // 1048576

  char* ws = (char*)d_ws;
  __bf16* xb  = (__bf16*)(ws);
  __bf16* wqb = (__bf16*)(ws + ((size_t)16 << 20));
  __bf16* wkb = (__bf16*)(ws + ((size_t)18 << 20));
  __bf16* wvb = (__bf16*)(ws + ((size_t)20 << 20));
  __bf16* wob = (__bf16*)(ws + ((size_t)22 << 20));
  __bf16* Qb  = (__bf16*)(ws + ((size_t)24 << 20));
  __bf16* Kb  = (__bf16*)(ws + ((size_t)40 << 20));
  __bf16* Vb  = (__bf16*)(ws + ((size_t)56 << 20));
  __bf16* Cb  = (__bf16*)(ws + ((size_t)72 << 20));

  cvt_f32_bf16<<<(nX + 255) / 256, 256, 0, stream>>>(x,  xb,  nX);
  cvt_f32_bf16<<<(nW + 255) / 256, 256, 0, stream>>>(Wq, wqb, nW);
  cvt_f32_bf16<<<(nW + 255) / 256, 256, 0, stream>>>(Wk, wkb, nW);
  cvt_f32_bf16<<<(nW + 255) / 256, 256, 0, stream>>>(Wv, wvb, nW);
  cvt_f32_bf16<<<(nW + 255) / 256, 256, 0, stream>>>(Wo, wob, nW);

  dim3 ggrid(E / GBN, M / GBM);   // (8, 64)
  gemm_bf16<<<ggrid, 128, 0, stream>>>(xb, wqb, bq, nullptr, Qb, M, E, E, 1);
  gemm_bf16<<<ggrid, 128, 0, stream>>>(xb, wkb, bk, nullptr, Kb, M, E, E, 1);
  gemm_bf16<<<ggrid, 128, 0, stream>>>(xb, wvb, bv, nullptr, Vb, M, E, E, 1);

  flash_attn<<<MHA_B * MHA_H * (MHA_S / 64), 128, 0, stream>>>(Qb, Kb, Vb, Cb);

  gemm_bf16<<<ggrid, 128, 0, stream>>>(Cb, wob, bo, out, nullptr, M, E, E, 0);
}
